// ProposalModule_72911364817658
// MI455X (gfx1250) — matrix operand order repaired and checked by reading, not compiled, not run on hardware
//
#include <hip/hip_runtime.h>

// ---------------------------------------------------------------------------
// ProposalModule for MI455X (gfx1250, wave32, WMMA).
// Heavy compute (grouped-point MLP + FC head) runs on v_wmma_f32_16x16x32_f16
// with fp32 accumulation. FPS / ball-query are latency-bound scalar kernels.
// ---------------------------------------------------------------------------

#define NB   32      // batch
#define NK   1024    // points
#define NCH  256     // feature channels
#define NP   256     // proposals
#define NS   16      // samples per ball
#define NPROP (NB*NP)            // 8192
#define K0PAD 288                // 259 padded to 9*32
#define RINV (1.0f/0.3f)
#define R2   0.09f

// output layout (floats), return order of the reference tuple
#define OFF_OBJ    0
#define OFF_CENTER 16384
#define OFF_HS     40960
#define OFF_HRN    139264
#define OFF_HR     237568
#define OFF_SS     335872
#define OFF_SRN    483328
#define OFF_SR     925696
#define OFF_SEM    1368064
#define OFF_NXYZ   1515520
#define OFF_INDS   1540096

// workspace layout (bytes)
#define WS_INDS    0          // 8192 i32
#define WS_NXYZ    32768      // 8192*3 f32
#define WS_IDX     131072     // 8192*16 i32
#define WS_H0      655360     // 128*288 f16  (w_mlp0^T, K-padded)
#define WS_H1      729088     // 128*128 f16
#define WS_H2      761856
#define WS_H3      794624
#define WS_H4      827392
#define WS_H5      860160     // w3^T padded N->128
#define WS_FEATH   892928     // 8192*128 f16 pooled features
// total ~2.99 MB

typedef __attribute__((ext_vector_type(16))) _Float16 v16h;
typedef __attribute__((ext_vector_type(8)))  _Float16 v8h;
typedef __attribute__((ext_vector_type(8)))  float    v8f;

union AF16 { v16h v; v8h h8[2]; _Float16 h[16]; };
union AF32 { v8f  v; float f[8]; };

// A fragment (16x32 f16, one M-row per lane, m = lane&15):
// elements e=0..15 -> K = kk + half*16 + e (contiguous run)
__device__ __forceinline__ v16h load_a(const _Float16* row, int kk, int half_id) {
  AF16 a;
  const _Float16* rp = row + kk + half_id * 16;
  a.h8[0] = *(const v8h*)(rp);
  a.h8[1] = *(const v8h*)(rp + 8);
  return a.v;
}

// B fragment (32x16 f16, one N-column per lane, n = lane&15), weights stored
// transposed wT[n][k]: elements e -> K = kk + half*16 + e (contiguous run).
__device__ __forceinline__ v16h load_b(const _Float16* wrow, int kk, int half_id) {
  AF16 b;
  const _Float16* rp = wrow + kk + half_id * 16;
  b.h8[0] = *(const v8h*)(rp);
  b.h8[1] = *(const v8h*)(rp + 8);
  return b.v;
}

// ---------------------------------------------------------------------------
// Weight prep: transpose + f16 convert + pad. grid=128 (n), block=288 (k).
// ---------------------------------------------------------------------------
__global__ void prep_kernel(const float* __restrict__ w0, const float* __restrict__ wm1,
                            const float* __restrict__ wm2, const float* __restrict__ wf1,
                            const float* __restrict__ wf2, const float* __restrict__ w3,
                            _Float16* __restrict__ h0, _Float16* __restrict__ h1,
                            _Float16* __restrict__ h2, _Float16* __restrict__ h3,
                            _Float16* __restrict__ h4, _Float16* __restrict__ h5) {
  int n = blockIdx.x, k = threadIdx.x;
  h0[n * K0PAD + k] = (k < 259) ? (_Float16)w0[k * 128 + n] : (_Float16)0.0f;
  if (k < 128) {
    h1[n * 128 + k] = (_Float16)wm1[k * 128 + n];
    h2[n * 128 + k] = (_Float16)wm2[k * 128 + n];
    h3[n * 128 + k] = (_Float16)wf1[k * 128 + n];
    h4[n * 128 + k] = (_Float16)wf2[k * 128 + n];
    h5[n * 128 + k] = (n < 119) ? (_Float16)w3[k * 119 + n] : (_Float16)0.0f;
  }
}

// ---------------------------------------------------------------------------
// Farthest point sampling. grid=B, block=256. Sequential scan of 255 dependent
// argmax steps; wave-shuffle reductions keep it to 2 barriers per step.
// ---------------------------------------------------------------------------
__global__ void __launch_bounds__(256) fps_kernel(const float* __restrict__ xyz,
                                                  int* __restrict__ inds_ws) {
  __shared__ float sx[NK], sy[NK], sz[NK], sd[NK];
  __shared__ float rv[8];
  __shared__ int   ri[8];
  __shared__ int   s_last;
  const int b = blockIdx.x, t = threadIdx.x;
  const int lane = t & 31, wv = t >> 5;
  for (int k = t; k < NK; k += 256) {
    sx[k] = xyz[(b * NK + k) * 3 + 0];
    sy[k] = xyz[(b * NK + k) * 3 + 1];
    sz[k] = xyz[(b * NK + k) * 3 + 2];
    sd[k] = 1e10f;
  }
  if (t == 0) { s_last = 0; inds_ws[b * NP + 0] = 0; }
  __syncthreads();
  for (int p = 1; p < NP; ++p) {
    const int last = s_last;
    const float lx = sx[last], ly = sy[last], lz = sz[last];
    float bv = -1.0f; int bi = 0;
    for (int k = t; k < NK; k += 256) {
      float dx = sx[k] - lx, dy = sy[k] - ly, dz = sz[k] - lz;
      float d = fminf(sd[k], dx * dx + dy * dy + dz * dz);
      sd[k] = d;
      if (d > bv) { bv = d; bi = k; }   // ascending k => first-max kept
    }
    // wave reduce (argmax, lower index on ties)
    for (int off = 16; off > 0; off >>= 1) {
      float ov = __shfl_xor(bv, off, 32);
      int   oi = __shfl_xor(bi, off, 32);
      if (ov > bv || (ov == bv && oi < bi)) { bv = ov; bi = oi; }
    }
    if (lane == 0) { rv[wv] = bv; ri[wv] = bi; }
    __syncthreads();
    if (t == 0) {
      float fv = rv[0]; int fi = ri[0];
      for (int i = 1; i < 8; ++i)
        if (rv[i] > fv || (rv[i] == fv && ri[i] < fi)) { fv = rv[i]; fi = ri[i]; }
      s_last = fi;
      inds_ws[b * NP + p] = fi;
    }
    __syncthreads();
  }
}

// ---------------------------------------------------------------------------
// Ball query + new_xyz + inds-to-output. grid=B, block=P(256), thread = p.
// Selects the first NS in-radius indices in ascending k order (reference
// semantics: sorted candidate indices, pad with first).
// ---------------------------------------------------------------------------
__global__ void __launch_bounds__(256) ball_kernel(const float* __restrict__ xyz,
                                                   const int* __restrict__ inds_ws,
                                                   int* __restrict__ idx_ws,
                                                   float* __restrict__ nxyz_ws,
                                                   float* __restrict__ out) {
  __shared__ float sx[NK], sy[NK], sz[NK];
  const int b = blockIdx.x, p = threadIdx.x;
  for (int k = p; k < NK; k += 256) {
    sx[k] = xyz[(b * NK + k) * 3 + 0];
    sy[k] = xyz[(b * NK + k) * 3 + 1];
    sz[k] = xyz[(b * NK + k) * 3 + 2];
  }
  __syncthreads();
  const int bp = b * NP + p;
  const int ind = inds_ws[bp];
  const float cx = sx[ind], cy = sy[ind], cz = sz[ind];
  nxyz_ws[bp * 3 + 0] = cx;  nxyz_ws[bp * 3 + 1] = cy;  nxyz_ws[bp * 3 + 2] = cz;
  out[OFF_NXYZ + bp * 3 + 0] = cx;
  out[OFF_NXYZ + bp * 3 + 1] = cy;
  out[OFF_NXYZ + bp * 3 + 2] = cz;
  out[OFF_INDS + bp] = (float)ind;
  int cnt = 0, first = NK - 1;
  for (int k = 0; k < NK; ++k) {
    float dx = sx[k] - cx, dy = sy[k] - cy, dz = sz[k] - cz;
    if (dx * dx + dy * dy + dz * dz < R2) {
      if (cnt == 0) first = k;
      if (cnt < NS) idx_ws[bp * NS + cnt] = k;
      ++cnt;
    }
  }
  for (int s = (cnt < NS ? cnt : NS); s < NS; ++s) idx_ws[bp * NS + s] = first;
}

// ---------------------------------------------------------------------------
// Fused grouped MLP (3 layers) + max-pool over NS. One workgroup (8 waves)
// per proposal: M-tile = its 16 samples, 8 waves cover N=128.
// Layer0: K=288 (259 padded), layers 1/2: K=128. f16 WMMA, f32 accumulate.
// ---------------------------------------------------------------------------
__global__ void __launch_bounds__(256) mlp_kernel(
    const float* __restrict__ xyz, const float* __restrict__ features,
    const int* __restrict__ idx_ws, const float* __restrict__ nxyz_ws,
    const _Float16* __restrict__ h0, const _Float16* __restrict__ h1,
    const _Float16* __restrict__ h2,
    const float* __restrict__ b0, const float* __restrict__ g0, const float* __restrict__ e0,
    const float* __restrict__ b1, const float* __restrict__ g1, const float* __restrict__ e1,
    const float* __restrict__ b2, const float* __restrict__ g2, const float* __restrict__ e2,
    _Float16* __restrict__ featH) {
  __shared__ int sidx[NS];
  __shared__ __align__(16) _Float16 x0[16 * K0PAD];
  __shared__ __align__(16) _Float16 x1[16 * 128];
  __shared__ __align__(16) _Float16 x2[16 * 128];
  const int bp = blockIdx.x;
  const int b  = bp >> 8;
  const int t  = threadIdx.x;
  if (t < NS) sidx[t] = idx_ws[bp * NS + t];
  __syncthreads();
  const float cx = nxyz_ws[bp * 3 + 0], cy = nxyz_ws[bp * 3 + 1], cz = nxyz_ws[bp * 3 + 2];
  for (int i = t; i < 16 * K0PAD; i += 256) {
    const int s = i / K0PAD, c = i - s * K0PAD;
    const int ks = sidx[s];
    float v = 0.0f;
    if (c < 3) {
      float pv = xyz[(b * NK + ks) * 3 + c];
      float ce = (c == 0) ? cx : ((c == 1) ? cy : cz);
      v = (pv - ce) * RINV;
    } else if (c < 259) {
      v = features[(b * NK + ks) * NCH + (c - 3)];
    }
    x0[s * K0PAD + c] = (_Float16)v;
  }
  __syncthreads();

  const int lane = t & 31, wv = t >> 5;
  const int half_id = lane >> 4, col = lane & 15;
  const int n = wv * 16 + col;

  // ---- layer 0 ----
  AF32 a0 = {};
  {
    const _Float16* arow = x0 + col * K0PAD;
    const _Float16* wrow = h0 + n * K0PAD;
    for (int kk = 0; kk < K0PAD; kk += 32) {
      v16h a = load_a(arow, kk, half_id);
      v16h w = load_b(wrow, kk, half_id);
      a0.v = __builtin_amdgcn_wmma_f32_16x16x32_f16(false, a, false, w, (short)0, a0.v, false, false);
    }
  }
  {
    const float gg = g0[n], bb = b0[n], ee = e0[n];
    for (int vv = 0; vv < 8; ++vv) {
      const int m = vv + half_id * 8;
      x1[m * 128 + n] = (_Float16)fmaxf(gg * (a0.f[vv] + bb) + ee, 0.0f);
    }
  }
  __syncthreads();

  // ---- layer 1 ----
  AF32 a1 = {};
  {
    const _Float16* arow = x1 + col * 128;
    const _Float16* wrow = h1 + n * 128;
    for (int kk = 0; kk < 128; kk += 32) {
      v16h a = load_a(arow, kk, half_id);
      v16h w = load_b(wrow, kk, half_id);
      a1.v = __builtin_amdgcn_wmma_f32_16x16x32_f16(false, a, false, w, (short)0, a1.v, false, false);
    }
  }
  {
    const float gg = g1[n], bb = b1[n], ee = e1[n];
    for (int vv = 0; vv < 8; ++vv) {
      const int m = vv + half_id * 8;
      x2[m * 128 + n] = (_Float16)fmaxf(gg * (a1.f[vv] + bb) + ee, 0.0f);
    }
  }
  __syncthreads();

  // ---- layer 2 + max-pool over the 16 samples (tile rows) ----
  AF32 a2 = {};
  {
    const _Float16* arow = x2 + col * 128;
    const _Float16* wrow = h2 + n * 128;
    for (int kk = 0; kk < 128; kk += 32) {
      v16h a = load_a(arow, kk, half_id);
      v16h w = load_b(wrow, kk, half_id);
      a2.v = __builtin_amdgcn_wmma_f32_16x16x32_f16(false, a, false, w, (short)0, a2.v, false, false);
    }
  }
  {
    const float gg = g2[n], bb = b2[n], ee = e2[n];
    float mx = 0.0f;                          // relu output >= 0
    for (int vv = 0; vv < 8; ++vv)
      mx = fmaxf(mx, fmaxf(gg * (a2.f[vv] + bb) + ee, 0.0f));
    mx = fmaxf(mx, __shfl_xor(mx, 16, 32));   // combine M=0..7 with M=8..15
    if (half_id == 0) featH[bp * 128 + n] = (_Float16)mx;
  }
}

// ---------------------------------------------------------------------------
// FC head: 16 proposals per workgroup, 2x (128x128 relu6) + 128x119, decode.
// ---------------------------------------------------------------------------
__global__ void __launch_bounds__(256) head_kernel(
    const _Float16* __restrict__ featH,
    const _Float16* __restrict__ h3, const _Float16* __restrict__ h4,
    const _Float16* __restrict__ h5,
    const float* __restrict__ b1, const float* __restrict__ g1, const float* __restrict__ e1,
    const float* __restrict__ b2, const float* __restrict__ g2, const float* __restrict__ e2,
    const float* __restrict__ b3, const float* __restrict__ msz,
    const float* __restrict__ nxyz_ws, float* __restrict__ out) {
  __shared__ __align__(16) _Float16 y1[16 * 128];
  __shared__ __align__(16) _Float16 y2[16 * 128];
  const int pr0 = blockIdx.x * 16;
  const int t = threadIdx.x, lane = t & 31, wv = t >> 5;
  const int half_id = lane >> 4, col = lane & 15;
  const int n = wv * 16 + col;

  // ---- FC1 (relu6) ----
  AF32 a1 = {};
  {
    const _Float16* arow = featH + (pr0 + col) * 128;
    const _Float16* wrow = h3 + n * 128;
    for (int kk = 0; kk < 128; kk += 32) {
      v16h a = load_a(arow, kk, half_id);
      v16h w = load_b(wrow, kk, half_id);
      a1.v = __builtin_amdgcn_wmma_f32_16x16x32_f16(false, a, false, w, (short)0, a1.v, false, false);
    }
  }
  {
    const float gg = g1[n], bb = b1[n], ee = e1[n];
    for (int vv = 0; vv < 8; ++vv) {
      const int m = vv + half_id * 8;
      float y = gg * (a1.f[vv] + bb) + ee;
      y1[m * 128 + n] = (_Float16)fminf(fmaxf(y, 0.0f), 6.0f);
    }
  }
  __syncthreads();

  // ---- FC2 (relu6) ----
  AF32 a2 = {};
  {
    const _Float16* arow = y1 + col * 128;
    const _Float16* wrow = h4 + n * 128;
    for (int kk = 0; kk < 128; kk += 32) {
      v16h a = load_a(arow, kk, half_id);
      v16h w = load_b(wrow, kk, half_id);
      a2.v = __builtin_amdgcn_wmma_f32_16x16x32_f16(false, a, false, w, (short)0, a2.v, false, false);
    }
  }
  {
    const float gg = g2[n], bb = b2[n], ee = e2[n];
    for (int vv = 0; vv < 8; ++vv) {
      const int m = vv + half_id * 8;
      float y = gg * (a2.f[vv] + bb) + ee;
      y2[m * 128 + n] = (_Float16)fminf(fmaxf(y, 0.0f), 6.0f);
    }
  }
  __syncthreads();

  // ---- FC3 (119 out, padded to 128) + decode ----
  AF32 a3 = {};
  {
    const _Float16* arow = y2 + col * 128;
    const _Float16* wrow = h5 + n * 128;
    for (int kk = 0; kk < 128; kk += 32) {
      v16h a = load_a(arow, kk, half_id);
      v16h w = load_b(wrow, kk, half_id);
      a3.v = __builtin_amdgcn_wmma_f32_16x16x32_f16(false, a, false, w, (short)0, a3.v, false, false);
    }
  }
  if (n < 119) {
    const float bb = b3[n];
    for (int vv = 0; vv < 8; ++vv) {
      const int pr = pr0 + vv + half_id * 8;
      const float val = a3.f[vv] + bb;
      if (n < 3) {
        out[OFF_CENTER + pr * 3 + n] = nxyz_ws[pr * 3 + n] + val;   // center = new_xyz + offset
      } else if (n < 5) {
        out[OFF_OBJ + pr * 2 + (n - 3)] = val;
      } else if (n < 17) {
        out[OFF_HS + pr * 12 + (n - 5)] = val;
      } else if (n < 35) {
        out[OFF_SS + pr * 18 + (n - 17)] = val;
      } else if (n < 47) {
        out[OFF_HRN + pr * 12 + (n - 35)] = val;
        out[OFF_HR  + pr * 12 + (n - 35)] = val * 0.26179938779914946f; // PI/NH
      } else if (n < 101) {
        const int q = n - 47;
        out[OFF_SRN + pr * 54 + q] = val;
        out[OFF_SR  + pr * 54 + q] = val * msz[q];                   // mean_size_arr flat (NC,3)
      } else {
        out[OFF_SEM + pr * 18 + (n - 101)] = val;
      }
    }
  }
}

// ---------------------------------------------------------------------------
extern "C" void kernel_launch(void* const* d_in, const int* in_sizes, int n_in,
                              void* d_out, int out_size, void* d_ws, size_t ws_size,
                              hipStream_t stream) {
  const float* xyz      = (const float*)d_in[0];
  const float* features = (const float*)d_in[1];
  const float* msz      = (const float*)d_in[2];
  const float* w_mlp0 = (const float*)d_in[3];
  const float* b_mlp0 = (const float*)d_in[4];
  const float* g_mlp0 = (const float*)d_in[5];
  const float* e_mlp0 = (const float*)d_in[6];
  const float* w_mlp1 = (const float*)d_in[7];
  const float* b_mlp1 = (const float*)d_in[8];
  const float* g_mlp1 = (const float*)d_in[9];
  const float* e_mlp1 = (const float*)d_in[10];
  const float* w_mlp2 = (const float*)d_in[11];
  const float* b_mlp2 = (const float*)d_in[12];
  const float* g_mlp2 = (const float*)d_in[13];
  const float* e_mlp2 = (const float*)d_in[14];
  const float* w1  = (const float*)d_in[15];
  const float* bf1 = (const float*)d_in[16];
  const float* gf1 = (const float*)d_in[17];
  const float* ef1 = (const float*)d_in[18];
  const float* w2  = (const float*)d_in[19];
  const float* bf2 = (const float*)d_in[20];
  const float* gf2 = (const float*)d_in[21];
  const float* ef2 = (const float*)d_in[22];
  const float* w3  = (const float*)d_in[23];
  const float* b3  = (const float*)d_in[24];

  char* ws = (char*)d_ws;
  int*      inds_ws = (int*)(ws + WS_INDS);
  float*    nxyz_ws = (float*)(ws + WS_NXYZ);
  int*      idx_ws  = (int*)(ws + WS_IDX);
  _Float16* h0 = (_Float16*)(ws + WS_H0);
  _Float16* h1 = (_Float16*)(ws + WS_H1);
  _Float16* h2 = (_Float16*)(ws + WS_H2);
  _Float16* h3 = (_Float16*)(ws + WS_H3);
  _Float16* h4 = (_Float16*)(ws + WS_H4);
  _Float16* h5 = (_Float16*)(ws + WS_H5);
  _Float16* featH = (_Float16*)(ws + WS_FEATH);
  float* out = (float*)d_out;

  prep_kernel<<<128, K0PAD, 0, stream>>>(w_mlp0, w_mlp1, w_mlp2, w1, w2, w3,
                                         h0, h1, h2, h3, h4, h5);
  fps_kernel<<<NB, 256, 0, stream>>>(xyz, inds_ws);
  ball_kernel<<<NB, 256, 0, stream>>>(xyz, inds_ws, idx_ws, nxyz_ws, out);
  mlp_kernel<<<NPROP, 256, 0, stream>>>(xyz, features, idx_ws, nxyz_ws,
                                        h0, h1, h2,
                                        b_mlp0, g_mlp0, e_mlp0,
                                        b_mlp1, g_mlp1, e_mlp1,
                                        b_mlp2, g_mlp2, e_mlp2,
                                        featH);
  head_kernel<<<NPROP / 16, 256, 0, stream>>>(featH, h3, h4, h5,
                                              bf1, gf1, ef1,
                                              bf2, gf2, ef2,
                                              b3, msz, nxyz_ws, out);
}
